// RNN_51187420233805
// MI455X (gfx1250) — matrix operand order repaired and checked by reading, not compiled, not run on hardware
//
#include <hip/hip_runtime.h>
#include <hip/hip_bf16.h>
#include <hip/hip_fp16.h>

// Problem dims (fixed by reference)
#define I_DIM 256
#define H_DIM 1024
#define B_DIM 256
#define T_DIM 256

// Persistent-scan tiling
#define NSTRIP 64                   // output columns per workgroup
#define NBLK   (H_DIM / NSTRIP)     // 16 workgroups in the scan grid
#define LDS_STRIDE 1032             // halfs per LDS row (1024 + 8 pad -> 4-dword skew)

typedef __attribute__((ext_vector_type(16))) _Float16 v16h;
typedef __attribute__((ext_vector_type(8)))  _Float16 v8h;
typedef __attribute__((ext_vector_type(8)))  float    v8f;
typedef __attribute__((ext_vector_type(4)))  int      v4i;

__device__ __forceinline__ float fast_tanh(float x) {
#if __has_builtin(__builtin_amdgcn_tanhf)
    return __builtin_amdgcn_tanhf(x);   // gfx1250 v_tanh_f32
#else
    return tanhf(x);
#endif
}

// ---- CDNA5 async global->LDS copy (16 bytes), with portable fallback -------
// Builtin signature (verified via hipcc diagnostic): (v4i* AS1 src, v4i* AS3 dst,
// imm offset, imm cpol), b128 payload.
__device__ __forceinline__ void async_copy16(const _Float16* g, _Float16* l) {
#if __has_builtin(__builtin_amdgcn_global_load_async_to_lds_b128)
    __builtin_amdgcn_global_load_async_to_lds_b128(
        (__attribute__((address_space(1))) v4i*)g,
        (__attribute__((address_space(3))) v4i*)l,
        0, 0);
#else
    *(v8h*)l = *(const v8h*)g;
#endif
}

__device__ __forceinline__ void wait_async_lds() {
#if __has_builtin(__builtin_amdgcn_global_load_async_to_lds_b128)
#if __has_builtin(__builtin_amdgcn_s_wait_asynccnt)
    __builtin_amdgcn_s_wait_asynccnt(0);
#else
    asm volatile("s_wait_asynccnt 0" ::: "memory");
#endif
#endif
}

// ---- device-scope barrier for the persistent scan (16 co-resident WGs) -----
__device__ __forceinline__ void grid_barrier(unsigned* cnt, unsigned target) {
    __threadfence();                 // release this thread's global stores
    __syncthreads();
    if (threadIdx.x == 0) {
        __hip_atomic_fetch_add(cnt, 1u, __ATOMIC_RELEASE, __HIP_MEMORY_SCOPE_AGENT);
        while (__hip_atomic_load(cnt, __ATOMIC_ACQUIRE, __HIP_MEMORY_SCOPE_AGENT) < target) {
            __builtin_amdgcn_s_sleep(2);
        }
    }
    __syncthreads();
}

// ---------------------------------------------------------------------------
// f32 -> f16 convert (weights)
// ---------------------------------------------------------------------------
__global__ void cvt_kernel(const float* __restrict__ in, _Float16* __restrict__ out, int n) {
    int i = blockIdx.x * blockDim.x + threadIdx.x;
    if (i < n) out[i] = (_Float16)in[i];
}

// x [B,T,I] f32  ->  x16 [T,B,I] f16   (coalesced along I)
__global__ void xpose_cvt_kernel(const float* __restrict__ x, _Float16* __restrict__ out) {
    int t = blockIdx.x, b = blockIdx.y, i = threadIdx.x;
    out[((size_t)t * B_DIM + b) * I_DIM + i] =
        (_Float16)x[((size_t)b * T_DIM + t) * I_DIM + i];
}

// ---------------------------------------------------------------------------
// Big parallel WMMA GEMM for the input projections:
//   out[M,N](f16) = A[M,K](f16) . W[N,K](f16)^T + bias0[n] + bias1[n]
// 8 waves (2x4), wave tile 64x32 (MT=4, NT=2), workgroup tile 128x128.
// Fragment layouts per CDNA5 ISA 7.12.2 (wave32).
// ---------------------------------------------------------------------------
template <int MT, int NT>
__global__ __launch_bounds__(256)
void gemm_bias_kernel(const _Float16* __restrict__ A,
                      const _Float16* __restrict__ W,
                      const float*    __restrict__ bias0,
                      const float*    __restrict__ bias1,
                      _Float16*       __restrict__ out,
                      int M, int N, int K) {
    const int lane = threadIdx.x & 31;
    const int wave = threadIdx.x >> 5;
    const int wr   = wave >> 2;       // 0..1
    const int wc   = wave & 3;        // 0..3
    const int lm   = lane & 15;
    const int sel  = lane >> 4;

    const int mW = (blockIdx.y * 2 + wr) * (MT * 16);
    const int nW = (blockIdx.x * 4 + wc) * (NT * 16);

    v8f acc[MT][NT];
#pragma unroll
    for (int i = 0; i < MT; ++i)
#pragma unroll
        for (int j = 0; j < NT; ++j)
#pragma unroll
            for (int r = 0; r < 8; ++r) acc[i][j][r] = 0.0f;

    for (int kc = 0; kc < K; kc += 32) {
        v16h af[MT];
#pragma unroll
        for (int i = 0; i < MT; ++i) {
            const _Float16* ap = A + (size_t)(mW + i * 16 + lm) * K + kc + sel * 8;
            v8h lo = *(const v8h*)(ap);
            v8h hi = *(const v8h*)(ap + 16);
            af[i] = __builtin_shufflevector(lo, hi, 0, 1, 2, 3, 4, 5, 6, 7,
                                                    8, 9, 10, 11, 12, 13, 14, 15);
        }
        v16h bf[NT];
#pragma unroll
        for (int j = 0; j < NT; ++j) {
            const _Float16* bp = W + (size_t)(nW + j * 16 + lm) * K + kc + sel * 8;
            v8h blo = *(const v8h*)(bp);
            v8h bhi = *(const v8h*)(bp + 16);
            bf[j] = __builtin_shufflevector(blo, bhi, 0, 1, 2, 3, 4, 5, 6, 7,
                                                      8, 9, 10, 11, 12, 13, 14, 15);
        }
#pragma unroll
        for (int i = 0; i < MT; ++i)
#pragma unroll
            for (int j = 0; j < NT; ++j)
                acc[i][j] = __builtin_amdgcn_wmma_f32_16x16x32_f16(
                    false, af[i], false, bf[j], (short)0, acc[i][j], false, false);
    }

#pragma unroll
    for (int i = 0; i < MT; ++i)
#pragma unroll
        for (int j = 0; j < NT; ++j)
#pragma unroll
            for (int r = 0; r < 8; ++r) {
                const int row = mW + i * 16 + sel * 8 + r;
                const int col = nW + j * 16 + lm;
                const size_t idx = (size_t)row * N + col;
                out[idx] = (_Float16)(acc[i][j][r] + bias0[col] + bias1[col]);
            }
}

// ---------------------------------------------------------------------------
// Persistent recurrent scan (one launch per layer):
//   for t in 0..T-1:  hall[t] = tanh(inp[t] + hall[t-1] . W^T)
// Grid: 16 WGs, each owns a 64-column strip of H for ALL 256 rows.
// Its W strip (64x1024 f16 = 128KB + pad) is staged into LDS ONCE via
// async global->LDS copies and reused for all 256 timesteps.
// 8 waves = 4(M) x 2(N); wave tile 64x32 = 4x2 accumulator tiles.
// inp[t] operands are preloaded into VGPRs BEFORE the k-loop so their
// latency hides under the 32-chunk WMMA loop; the epilogue is pure VALU.
// Steps are separated by a device-scope atomic barrier (16 WGs co-resident).
// ---------------------------------------------------------------------------
__global__ __launch_bounds__(256)
void rnn_scan_kernel(const _Float16* __restrict__ Wg,    // [H,H] f16 row-major
                     const _Float16* __restrict__ inp,   // [T,B,H] f16
                     _Float16*       __restrict__ hall,  // [T,B,H] f16 chain
                     unsigned*       __restrict__ bar,
                     int T) {
    extern __shared__ _Float16 ldsW[];   // NSTRIP x LDS_STRIDE halfs

    const int tid  = threadIdx.x;
    const int lane = tid & 31;
    const int wave = tid >> 5;
    const int lm   = lane & 15;
    const int sel  = lane >> 4;
    const int wrow = wave & 3;           // 4 wave-groups in M
    const int wcol = wave >> 2;          // 2 wave-groups in N
    const int n0   = blockIdx.x * NSTRIP;

    // ---- stage W strip into LDS (async DMA path), once per layer ----
    for (int e = tid; e < NSTRIP * (H_DIM / 8); e += 256) {
        const int row = e >> 7;          // H/8 = 128 16-byte chunks per row
        const int c   = e & 127;
        async_copy16(Wg + (size_t)(n0 + row) * H_DIM + c * 8,
                     &ldsW[row * LDS_STRIDE + c * 8]);
    }
    wait_async_lds();
    __syncthreads();

    const int mW = wrow * 64;            // wave M origin (covers rows mW..mW+63)
    const int nL = wcol * 32;            // wave N origin inside the strip

    for (int t = 0; t < T; ++t) {
        const _Float16* inpT = inp + (size_t)t * B_DIM * H_DIM;

        // Preload inp[t] fragment into registers: 64 u16 loads issued here,
        // consumed only after the k-loop -> latency hidden by the GEMM.
        _Float16 inps[4][2][8];
#pragma unroll
        for (int i = 0; i < 4; ++i)
#pragma unroll
            for (int j = 0; j < 2; ++j)
#pragma unroll
                for (int r = 0; r < 8; ++r) {
                    const int row = mW + i * 16 + sel * 8 + r;
                    const int col = n0 + nL + j * 16 + lm;
                    inps[i][j][r] = inpT[(size_t)row * H_DIM + col];
                }

        // Warm L2 for the NEXT timestep's inp fragment.
        {
            const int tn = (t + 1 < T) ? (t + 1) : t;
            const _Float16* inpN = inp + (size_t)tn * B_DIM * H_DIM;
#pragma unroll
            for (int i = 0; i < 4; ++i)
                __builtin_prefetch(&inpN[(size_t)(mW + i * 16 + sel * 8) * H_DIM + n0 + nL + lm], 0, 1);
        }

        v8f acc[4][2];
#pragma unroll
        for (int i = 0; i < 4; ++i)
#pragma unroll
            for (int j = 0; j < 2; ++j)
#pragma unroll
                for (int r = 0; r < 8; ++r) acc[i][j][r] = 0.0f;

        if (t > 0) {
            const _Float16* hprev = hall + (size_t)(t - 1) * B_DIM * H_DIM;
            for (int kc = 0; kc < H_DIM; kc += 32) {
                v16h af[4];
#pragma unroll
                for (int i = 0; i < 4; ++i) {
                    const _Float16* ap = hprev + (size_t)(mW + i * 16 + lm) * H_DIM + kc + sel * 8;
                    v8h lo = *(const v8h*)(ap);
                    v8h hi = *(const v8h*)(ap + 16);
                    af[i] = __builtin_shufflevector(lo, hi, 0, 1, 2, 3, 4, 5, 6, 7,
                                                            8, 9, 10, 11, 12, 13, 14, 15);
                }
                v16h bf[2];
#pragma unroll
                for (int j = 0; j < 2; ++j) {
                    const int base = (nL + j * 16 + lm) * LDS_STRIDE + kc + sel * 8;
                    v8h lo = *(const v8h*)&ldsW[base];      // ds_load_b128
                    v8h hi = *(const v8h*)&ldsW[base + 16];
                    bf[j] = __builtin_shufflevector(lo, hi, 0, 1, 2, 3, 4, 5, 6, 7,
                                                            8, 9, 10, 11, 12, 13, 14, 15);
                }
#pragma unroll
                for (int i = 0; i < 4; ++i)
#pragma unroll
                    for (int j = 0; j < 2; ++j)
                        acc[i][j] = __builtin_amdgcn_wmma_f32_16x16x32_f16(
                            false, af[i], false, bf[j], (short)0, acc[i][j], false, false);
            }
        }

        _Float16* hout = hall + (size_t)t * B_DIM * H_DIM;
#pragma unroll
        for (int i = 0; i < 4; ++i)
#pragma unroll
            for (int j = 0; j < 2; ++j)
#pragma unroll
                for (int r = 0; r < 8; ++r) {
                    const int row = mW + i * 16 + sel * 8 + r;
                    const int col = n0 + nL + j * 16 + lm;
                    hout[(size_t)row * H_DIM + col] =
                        (_Float16)fast_tanh(acc[i][j][r] + (float)inps[i][j][r]);
                }

        if (t != T - 1) grid_barrier(bar, (unsigned)(NBLK * (t + 1)));
    }
}

// ---------------------------------------------------------------------------
// Final FC (O=1) + ReLU: out[b] = relu(dot(h[b,:], w) + b0)
// ---------------------------------------------------------------------------
__global__ void fc_relu_kernel(const _Float16* __restrict__ h,
                               const float* __restrict__ w,
                               const float* __restrict__ b,
                               float* __restrict__ out, int K) {
    const int row = threadIdx.x;  // 256 rows, one block
    float s = 0.0f;
    for (int k = 0; k < K; ++k) s += (float)h[(size_t)row * K + k] * w[k];
    s += b[0];
    out[row] = fmaxf(s, 0.0f);
}

// ---------------------------------------------------------------------------
extern "C" void kernel_launch(void* const* d_in, const int* in_sizes, int n_in,
                              void* d_out, int out_size, void* d_ws, size_t ws_size,
                              hipStream_t stream) {
    (void)in_sizes; (void)n_in; (void)out_size; (void)ws_size;

    const float* x    = (const float*)d_in[0];
    const float* Wih0 = (const float*)d_in[1];
    const float* Whh0 = (const float*)d_in[2];
    const float* bih0 = (const float*)d_in[3];
    const float* bhh0 = (const float*)d_in[4];
    const float* Wih1 = (const float*)d_in[5];
    const float* Whh1 = (const float*)d_in[6];
    const float* bih1 = (const float*)d_in[7];
    const float* bhh1 = (const float*)d_in[8];
    const float* fcW  = (const float*)d_in[9];
    const float* fcb  = (const float*)d_in[10];

    const size_t TB = (size_t)T_DIM * B_DIM;   // 65536

    // Workspace carve (256B aligned chunks; ~310 MB total)
    char* p = (char*)d_ws;
    auto carve = [&](size_t bytes) -> char* {
        char* r = p;
        p += (bytes + 255) & ~(size_t)255;
        return r;
    };
    _Float16* x16  = (_Float16*)carve(TB * I_DIM * 2);
    _Float16* wih0 = (_Float16*)carve((size_t)H_DIM * I_DIM * 2);
    _Float16* whh0 = (_Float16*)carve((size_t)H_DIM * H_DIM * 2);
    _Float16* wih1 = (_Float16*)carve((size_t)H_DIM * H_DIM * 2);
    _Float16* whh1 = (_Float16*)carve((size_t)H_DIM * H_DIM * 2);
    _Float16* inp  = (_Float16*)carve(TB * H_DIM * 2);
    _Float16* hall = (_Float16*)carve(TB * H_DIM * 2);
    unsigned* bar  = (unsigned*)carve(256);

    const size_t scan_lds = (size_t)NSTRIP * LDS_STRIDE * sizeof(_Float16); // ~129 KB

    // 1) convert / transpose inputs to fp16
    xpose_cvt_kernel<<<dim3(T_DIM, B_DIM), I_DIM, 0, stream>>>(x, x16);
    {
        int n0 = H_DIM * I_DIM;
        cvt_kernel<<<(n0 + 255) / 256, 256, 0, stream>>>(Wih0, wih0, n0);
        int n1 = H_DIM * H_DIM;
        cvt_kernel<<<(n1 + 255) / 256, 256, 0, stream>>>(Whh0, whh0, n1);
        cvt_kernel<<<(n1 + 255) / 256, 256, 0, stream>>>(Wih1, wih1, n1);
        cvt_kernel<<<(n1 + 255) / 256, 256, 0, stream>>>(Whh1, whh1, n1);
    }

    // 2) Layer-0 input projection: inp = x16 . Wih0^T + (b_ih0 + b_hh0)
    gemm_bias_kernel<4, 2><<<dim3(8, 512), 256, 0, stream>>>(
        x16, wih0, bih0, bhh0, inp, (int)TB, H_DIM, I_DIM);

    // 3) Layer-0 recurrent scan (persistent, 1 launch for all 256 steps)
    (void)hipMemsetAsync(bar, 0, sizeof(unsigned), stream);
    rnn_scan_kernel<<<dim3(NBLK), 256, scan_lds, stream>>>(whh0, inp, hall, bar, T_DIM);

    // 4) Layer-1 input projection: inp = hall . Wih1^T + (b_ih1 + b_hh1)
    gemm_bias_kernel<4, 2><<<dim3(8, 512), 256, 0, stream>>>(
        hall, wih1, bih1, bhh1, inp, (int)TB, H_DIM, H_DIM);

    // 5) Layer-1 recurrent scan (reuses hall storage; only h[T-1] consumed)
    (void)hipMemsetAsync(bar, 0, sizeof(unsigned), stream);
    rnn_scan_kernel<<<dim3(NBLK), 256, scan_lds, stream>>>(whh1, inp, hall, bar, T_DIM);

    // 6) FC (O=1) + ReLU on the last hidden state
    fc_relu_kernel<<<1, B_DIM, 0, stream>>>(
        hall + (size_t)(T_DIM - 1) * B_DIM * H_DIM, fcW, fcb, (float*)d_out, H_DIM);
}
